// ArcFaceLoss_78185584657059
// MI455X (gfx1250) — matrix operand order repaired and checked by reading, not compiled, not run on hardware
//
#include <hip/hip_runtime.h>
#include <hip/hip_bf16.h>
#include <math.h>

typedef _Float16 v16h __attribute__((ext_vector_type(16)));
typedef _Float16 v8h  __attribute__((ext_vector_type(8)));
typedef _Float16 h4   __attribute__((ext_vector_type(4)));
typedef float    v8f  __attribute__((ext_vector_type(8)));
typedef float    f4   __attribute__((ext_vector_type(4)));

#define BATCH   1024
#define EMB     512
#define NCLS    100000
#define BM      128                      // rows per workgroup (8 waves x 16)
#define BN      128                      // cols per workgroup (8 16x16 tiles per wave)
#define KC      64                       // k-chunk staged in LDS
#define NCH     (EMB / KC)               // 8 chunks
#define NT      ((NCLS + BN - 1) / BN)   // 782 N-tiles
#define LDH     72                       // LDS row stride in halves (144 B: 16B-aligned, bank-spread)
#define SCALE_F 30.0f
#define COS_M   0.95533648912561f        // cos(0.3)
#define SIN_M   0.29552020666134f        // sin(0.3)
#define TH_C    (-0.95533648912561f)     // cos(pi-0.3)
#define MM_C    0.08865606199840f        // sin(pi-0.3)*0.3

// ---------------------------------------------------------------------------
// K0: embeddings f32 -> f16 (A operand). 1 MB, stays L2-resident.
// ---------------------------------------------------------------------------
__global__ void k_cvt_f16(const float* __restrict__ src, _Float16* __restrict__ dst) {
    int i = (blockIdx.x * blockDim.x + threadIdx.x) * 4;
    if (i < BATCH * EMB) {
        f4 v = *(const f4*)(src + i);
        h4 o = { (_Float16)v[0], (_Float16)v[1], (_Float16)v[2], (_Float16)v[3] };
        *(h4*)(dst + i) = o;
    }
}

// ---------------------------------------------------------------------------
// K1: label-column cosine in full f32 (1024 dot products). One wave per row.
// ---------------------------------------------------------------------------
__global__ void k_label(const float* __restrict__ E, const float* __restrict__ W,
                        const int* __restrict__ labels, float* __restrict__ coslam) {
    int lane = threadIdx.x & 31;
    int row  = (blockIdx.x * blockDim.x + threadIdx.x) >> 5;
    if (row >= BATCH) return;
    int lab = labels[row];
    lab = lab < 0 ? 0 : (lab >= NCLS ? NCLS - 1 : lab);
    const float* e = E + (size_t)row * EMB + lane * 16;
    const float* w = W + (size_t)lab * EMB + lane * 16;
    float dot = 0.f, ss = 0.f;
#pragma unroll
    for (int q = 0; q < 4; ++q) {
        f4 ev = *(const f4*)(e + q * 4);
        f4 wv = *(const f4*)(w + q * 4);
#pragma unroll
        for (int t = 0; t < 4; ++t) { dot += ev[t] * wv[t]; ss += wv[t] * wv[t]; }
    }
#pragma unroll
    for (int m = 1; m < 32; m <<= 1) { dot += __shfl_xor(dot, m, 32); ss += __shfl_xor(ss, m, 32); }
    if (lane == 0) coslam[row] = dot * rsqrtf(ss);
}

// ---------------------------------------------------------------------------
// K2: fused WMMA GEMM + weight-norm + online-softmax partials.
//   - B (f32 weight) cooperatively loaded (coalesced b128), converted to f16
//     ONCE per block, ds_store'd into a padded double-buffered LDS tile.
//   - A (f16) staged global->LDS with gfx1250 async-LDS loads (ASYNCcnt).
//   - All 8 B fragments preloaded into independent registers so the compiler
//     can hide LDS latency with partial s_wait_dscnt instead of 0-waits.
// ---------------------------------------------------------------------------
__global__ __launch_bounds__(256) void k_gemm(const _Float16* __restrict__ A,
                                              const float* __restrict__ W,
                                              float* __restrict__ pmax,
                                              float* __restrict__ psum) {
    __shared__ __align__(16) _Float16 sA[2][BM * LDH];   // 2 x 18 KB
    __shared__ __align__(16) _Float16 sB[2][BN * LDH];   // 2 x 18 KB
    __shared__ float sInv[BN];

    const int tid   = threadIdx.x;
    const int lane  = tid & 31;
    const int lhalf = (lane >> 4) & 1;
    const int lm    = lane & 15;
    const int wave  = tid >> 5;
    const int m0b   = blockIdx.x * BM;
    const int n0    = blockIdx.y * BN;

    // staging decomposition: group g = tid>>4 (0..15), segoff = tid&15
    const int sgrp = tid >> 4;
    const int soff = tid & 15;

    v8f   acc[8]  = {};
    float ssq8[8] = {};
    f4    breg[8];

    // A async stage for chunk kk into buffer nb: 128 rows x 64 halves (16 KB)
#define STAGE_A_ASYNC(kk, nb)                                                      \
    {                                                                              \
        _Pragma("unroll")                                                          \
        for (int i = 0; i < 4; ++i) {                                              \
            int seg = i * 256 + tid;                                               \
            int r = seg >> 3, so8 = seg & 7;                                       \
            const _Float16* gp = A + (size_t)(m0b + r) * EMB + (kk) + so8 * 8;     \
            unsigned ldsoff = (unsigned)(size_t)(&sA[nb][r * LDH + so8 * 8]);      \
            asm volatile("global_load_async_to_lds_b128 %0, %1, off"               \
                         :: "v"(ldsoff), "v"((unsigned long long)(size_t)gp)       \
                         : "memory");                                              \
        }                                                                          \
    }

    // B global loads for chunk kk (coalesced: 16 consecutive lanes = one column run)
#define LOAD_B(kk)                                                                 \
    {                                                                              \
        _Pragma("unroll")                                                          \
        for (int i = 0; i < 8; ++i) {                                              \
            int colL = i * 16 + sgrp;                                              \
            int cg = n0 + colL;                                                    \
            cg = cg < NCLS ? cg : NCLS - 1;                                        \
            breg[i] = *(const f4*)(W + (size_t)cg * EMB + (kk) + soff * 4);        \
        }                                                                          \
    }

    // convert + sum-of-squares + ds_store into buffer nb
#define CVT_STORE_B(nb)                                                            \
    {                                                                              \
        _Pragma("unroll")                                                          \
        for (int i = 0; i < 8; ++i) {                                              \
            f4 v = breg[i];                                                        \
            ssq8[i] += v[0]*v[0] + v[1]*v[1] + v[2]*v[2] + v[3]*v[3];              \
            h4 hv = { (_Float16)v[0], (_Float16)v[1], (_Float16)v[2], (_Float16)v[3] }; \
            int colL = i * 16 + sgrp;                                              \
            *(h4*)(&sB[nb][colL * LDH + soff * 4]) = hv;                           \
        }                                                                          \
    }

#define WAIT_ASYNC() asm volatile("s_wait_asynccnt 0x0" ::: "memory")

    // ---- prologue: stage chunk 0 ----
    STAGE_A_ASYNC(0, 0);
    LOAD_B(0);
    CVT_STORE_B(0);
    WAIT_ASYNC();
    __syncthreads();

    // ---- main loop over k-chunks, double buffered ----
    for (int c = 0; c < NCH; ++c) {
        const int cb = c & 1;
        const int nb = cb ^ 1;
        if (c + 1 < NCH) {
            STAGE_A_ASYNC((c + 1) * KC, nb);   // async A for next chunk
            LOAD_B((c + 1) * KC);              // B global loads in flight
        }

        // compute chunk c from LDS
        const _Float16* Aba = &sA[cb][0];
        const _Float16* Bba = &sB[cb][0];
#pragma unroll
        for (int ks = 0; ks < KC; ks += 32) {
            // A fragment (16-bit A 16x32): half0 K{ks..+7, ks+16..+23}, half1 +8
            const _Float16* ap = Aba + (wave * 16 + lm) * LDH + ks + (lhalf ? 8 : 0);
            v8h alo = *(const v8h*)ap;
            v8h ahi = *(const v8h*)(ap + 16);
            v16h a  = __builtin_shufflevector(alo, ahi, 0,1,2,3,4,5,6,7,8,9,10,11,12,13,14,15);

            // preload ALL 8 B fragments into independent registers first,
            // so ds_load latency is amortized across the WMMA sequence
            v16h bf[8];
#pragma unroll
            for (int j = 0; j < 8; ++j) {
                const _Float16* bp = Bba + (j * 16 + lm) * LDH + ks + (lhalf ? 8 : 0);
                v8h blo = *(const v8h*)bp;
                v8h bhi = *(const v8h*)(bp + 16);
                bf[j] = __builtin_shufflevector(blo, bhi, 0,1,2,3,4,5,6,7,8,9,10,11,12,13,14,15);
            }
#pragma unroll
            for (int j = 0; j < 8; ++j) {
                acc[j] = __builtin_amdgcn_wmma_f32_16x16x32_f16(
                    false, a, false, bf[j], (short)0, acc[j], false, false);
            }
        }

        if (c + 1 < NCH) {
            CVT_STORE_B(nb);                   // finish staging next chunk
            WAIT_ASYNC();
        }
        __syncthreads();
    }

    // ---- column inverse norms (cooperative): reduce 16-lane groups ----
#pragma unroll
    for (int i = 0; i < 8; ++i) {
        float v = ssq8[i];
#pragma unroll
        for (int m = 1; m < 16; m <<= 1) v += __shfl_xor(v, m, 32);
        if (soff == 0) sInv[i * 16 + sgrp] = rsqrtf(v);
    }
    __syncthreads();

    float invn[8];
#pragma unroll
    for (int j = 0; j < 8; ++j) invn[j] = sInv[j * 16 + lm];

    // ---- epilogue: per-row tile max + sum(exp) over this wave's 128 cols ----
    const int m0 = m0b + wave * 16;
#pragma unroll
    for (int i = 0; i < 8; ++i) {
        float x[8];
        float mx = -3.0e38f;
#pragma unroll
        for (int j = 0; j < 8; ++j) {
            bool valid = (n0 + j * 16) < NCLS;   // NCLS % 16 == 0
            float v = valid ? (SCALE_F * acc[j][i] * invn[j]) : -3.0e38f;
            x[j] = v;
            mx = fmaxf(mx, v);
        }
#pragma unroll
        for (int m = 1; m < 16; m <<= 1) mx = fmaxf(mx, __shfl_xor(mx, m, 32));
        float se = 0.f;
#pragma unroll
        for (int j = 0; j < 8; ++j)
            if ((n0 + j * 16) < NCLS) se += __expf(x[j] - mx);
#pragma unroll
        for (int m = 1; m < 16; m <<= 1) se += __shfl_xor(se, m, 32);
        if (lm == 0) {
            int row = m0 + i + (lhalf ? 8 : 0);
            size_t idx = (size_t)blockIdx.y * BATCH + row;
            pmax[idx] = mx;
            psum[idx] = se;
        }
    }
}

// ---------------------------------------------------------------------------
// K3: merge (max,sum) across tiles per row, ArcFace margin, per-row NLL.
// ---------------------------------------------------------------------------
__global__ void k_reduce(const float* __restrict__ pmax, const float* __restrict__ psum,
                         const float* __restrict__ coslam, float* __restrict__ nll) {
    int lane = threadIdx.x & 31;
    int row  = (blockIdx.x * blockDim.x + threadIdx.x) >> 5;
    if (row >= BATCH) return;
    float mx = -3.0e38f;
    for (int t = lane; t < NT; t += 32) mx = fmaxf(mx, pmax[(size_t)t * BATCH + row]);
#pragma unroll
    for (int m = 1; m < 32; m <<= 1) mx = fmaxf(mx, __shfl_xor(mx, m, 32));
    float s = 0.f;
    for (int t = lane; t < NT; t += 32) {
        size_t i = (size_t)t * BATCH + row;
        s += psum[i] * __expf(pmax[i] - mx);
    }
#pragma unroll
    for (int m = 1; m < 32; m <<= 1) s += __shfl_xor(s, m, 32);
    if (lane == 0) {
        float c    = coslam[row];
        float s2   = fminf(fmaxf(1.f - c * c, 0.f), 1.f);
        float sine = sqrtf(s2);
        float phi  = c * COS_M - sine * SIN_M;
        phi = (c > TH_C) ? phi : (c - MM_C);
        float xm = SCALE_F * c, pm = SCALE_F * phi;
        float tot = s - __expf(xm - mx) + __expf(pm - mx);   // swap label term for phi term
        nll[row] = mx + __logf(tot) - pm;                    // LSE - s*phi
    }
}

// ---------------------------------------------------------------------------
// K4: deterministic mean over the batch -> scalar.
// ---------------------------------------------------------------------------
__global__ void k_final(const float* __restrict__ nll, float* __restrict__ out) {
    __shared__ float sm[256];
    float v = 0.f;
    for (int i = threadIdx.x; i < BATCH; i += 256) v += nll[i];
    sm[threadIdx.x] = v;
    __syncthreads();
    for (int off = 128; off > 0; off >>= 1) {
        if (threadIdx.x < off) sm[threadIdx.x] += sm[threadIdx.x + off];
        __syncthreads();
    }
    if (threadIdx.x == 0) out[0] = sm[0] * (1.0f / BATCH);
}

// ---------------------------------------------------------------------------
extern "C" void kernel_launch(void* const* d_in, const int* in_sizes, int n_in,
                              void* d_out, int out_size, void* d_ws, size_t ws_size,
                              hipStream_t stream) {
    (void)in_sizes; (void)n_in; (void)out_size; (void)ws_size;
    const float* emb    = (const float*)d_in[0];   // [1024,512] f32
    const int*   labels = (const int*)d_in[1];     // [1024] int
    const float* weight = (const float*)d_in[2];   // [100000,512] f32

    char* ws = (char*)d_ws;
    _Float16* embF16 = (_Float16*)(ws);                       // 1 MB
    float* coslam    = (float*)(ws + (1u << 20));             // 4 KB
    float* nll       = (float*)(ws + (1u << 20) + 4096);      // 4 KB
    float* pmax      = (float*)(ws + (1u << 20) + 8192);      // NT*1024*4 = 3.2 MB
    float* psum      = (float*)(ws + (1u << 20) + 8192 + (size_t)NT * BATCH * 4);

    k_cvt_f16<<<(BATCH * EMB / 4 + 255) / 256, 256, 0, stream>>>(emb, embF16);
    k_label<<<BATCH / 8, 256, 0, stream>>>(emb, weight, labels, coslam);
    dim3 g(BATCH / BM, NT);   // x (M-tiles) fastest: blocks sharing a weight tile co-scheduled
    k_gemm<<<g, 256, 0, stream>>>(embF16, weight, pmax, psum);
    k_reduce<<<BATCH / 8, 256, 0, stream>>>(pmax, psum, coslam, nll);
    k_final<<<1, 256, 0, stream>>>(nll, (float*)d_out);
}
